// FNPGNN_Ablation_38019050504255
// MI455X (gfx1250) — compile-verified, matrix operand order checked by reading, not run on hardware
//
#include <hip/hip_runtime.h>
#include <hip/hip_bf16.h>
#include <math.h>

// =====================================================================
// FNP-GNN forward for MI455X (gfx1250, wave32, WMMA).
// One workgroup (256 thr = 8 waves) per batch element; h / Wh / adj /
// WMMA fragment buffers all resident in LDS (~260 KB of the 320 KB WGP
// LDS). All GEMMs use v_wmma_f32_16x16x32_f16 (f16 in, f32 accum).
// =====================================================================

typedef __attribute__((ext_vector_type(16))) _Float16 v16h;
typedef __attribute__((ext_vector_type(8)))  float    v8f;

#define NROI 116
#define NCH  128
#define CP   132          // padded f32 row stride (bank-conflict friendly)
#define NTHR 256

__device__ __forceinline__ float elu_f(float x) { return x > 0.f ? x : expm1f(x); }

// ISA 16-bit A-matrix 16x32 layout (CDNA5 §7.12.2), expressed as:
// element (m in 0..15, k in 0..31) -> index into a [32 lanes][16 halfs] tile.
// B (32x16) assumed transpose-symmetric (n plays the role of m).
__device__ __forceinline__ int frag_slot(int m, int k) {
  int p    = k >> 1;                         // K pair index 0..15
  int lane = (m & 15) | (((p >> 2) & 1) << 4);
  int v    = (p & 3)  | (((p >> 3) & 1) << 2);
  return lane * 16 + (v << 1) + (k & 1);
}

struct Params {
  const float *x, *adj;
  const float *enc_W, *enc_b;
  const float *bn_in_g, *bn_in_b, *bn_in_rm, *bn_in_rv;
  const float *vne;
  const float *gat_W[3], *gat_asrc[3], *gat_adst[3];
  const float *gat_g[3], *gat_b[3], *gat_rm[3], *gat_rv[3];
  const float *vnu_W, *vnu_b;
  const float *attn_W1, *attn_b1, *attn_W2, *attn_b2;
  const float *ln_g, *ln_b;
  float *flat_out;   // d_out + 1024*2
};

__constant__ int NET_OFF[10] = {0, 10, 16, 28, 34, 40, 46, 50, 58, 84};
__constant__ int NET_IDX[84] = {
  34,35,66,67,64,65,22,23,24,25,              // DMN
  0,1,56,57,68,69,                            // SMN
  42,43,44,45,46,47,48,49,50,51,52,53,        // VN
  28,29,30,31,32,33,                          // SN
  6,7,58,59,60,61,                            // FPN
  36,37,38,39,40,41,                          // LN
  10,11,14,15,                                // VAN
  70,71,72,73,74,75,76,77,                    // BGN
  90,91,92,93,94,95,96,97,98,99,100,101,102,103,104,105,
  106,107,108,109,110,111,112,113,114,115     // CereN
};

// 128x128x128 GEMM: wave w owns M-tile w; acc[nt] for nt=0..7.
__device__ __forceinline__ void gemm128(const _Float16* sA, const _Float16* sB, v8f* acc) {
  const int wave = threadIdx.x >> 5, lane = threadIdx.x & 31;
  v8f z = {};
#pragma unroll
  for (int q = 0; q < 8; ++q) acc[q] = z;
#pragma unroll
  for (int kt = 0; kt < 4; ++kt) {
    v16h a = *(const v16h*)(sA + (wave * 4 + kt) * 512 + lane * 16);
#pragma unroll
    for (int nt = 0; nt < 8; ++nt) {
      v16h b = *(const v16h*)(sB + (kt * 8 + nt) * 512 + lane * 16);
      acc[nt] = __builtin_amdgcn_wmma_f32_16x16x32_f16(
          false, a, false, b, (short)0, acc[nt], false, false);
    }
  }
}

__device__ __forceinline__ void restageA(const float* sh, _Float16* sAfrag) {
  for (int t = threadIdx.x; t < NCH * NCH; t += NTHR) {
    int m = t >> 7, k = t & 127;
    float v = (m < NROI) ? sh[m * CP + k] : 0.f;
    sAfrag[((m >> 4) * 4 + (k >> 5)) * 512 + frag_slot(m & 15, k & 31)] = (_Float16)v;
  }
}

__device__ __forceinline__ void colmean(const float* sh, float* sMean) {
  int tid = threadIdx.x;
  if (tid < NCH) {
    float s = 0.f;
    for (int i = 0; i < NROI; ++i) s += sh[i * CP + tid];
    sMean[tid] = s * (1.f / NROI);
  }
}

__device__ __forceinline__ void gat_layer(const Params& p, int l, float* sh, float* sWh,
                                          _Float16* sAfrag, _Float16* sBfrag,
                                          const float* sAdj, float* sSrc, float* sDst) {
  const int tid = threadIdx.x, wave = tid >> 5, lane = tid & 31;

  // ---- stage W_l as B fragments: B(k,n) = W[n][k]
  const float* W = p.gat_W[l];
  for (int t = tid; t < NCH * NCH; t += NTHR) {
    int n = t >> 7, k = t & 127;
    sBfrag[((k >> 5) * 8 + (n >> 4)) * 512 + frag_slot(n & 15, k & 31)] = (_Float16)W[n * NCH + k];
  }
  __syncthreads();

  // ---- Wh = h @ W^T (WMMA) -> sWh (f32)
  {
    v8f acc[8];
    gemm128(sAfrag, sBfrag, acc);
#pragma unroll
    for (int q = 0; q < 8; ++q)
#pragma unroll
      for (int r = 0; r < 8; ++r) {
        int row = wave * 16 + r + ((lane >> 4) << 3);
        int col = q * 16 + (lane & 15);
        sWh[row * CP + col] = acc[q][r];
      }
  }
  __syncthreads();

  // ---- restage B fragments from Wh (K = node j); compute src/dst
  for (int t = tid; t < NCH * NCH; t += NTHR) {
    int k = t >> 7, n = t & 127;
    float v = (k < NROI) ? sWh[k * CP + n] : 0.f;
    sBfrag[((k >> 5) * 8 + (n >> 4)) * 512 + frag_slot(n & 15, k & 31)] = (_Float16)v;
  }
  const float* asrc = p.gat_asrc[l];
  const float* adst = p.gat_adst[l];
  for (int t = tid; t < NROI * 4; t += NTHR) {
    int i = t >> 2, hd = t & 3;
    const float* row = sWh + i * CP + hd * 32;
    float s = 0.f, d = 0.f;
    for (int u = 0; u < 32; ++u) { float w = row[u]; s = fmaf(w, asrc[u], s); d = fmaf(w, adst[u], d); }
    sSrc[t] = s; sDst[t] = d;
  }
  __syncthreads();

  // ---- per head: masked softmax -> alpha (f16 A frags), WMMA aggregation
  for (int hd = 0; hd < 4; ++hd) {
    for (int i = wave; i < NCH; i += 8) {
      if (i >= NROI) {   // zero padded rows so WMMA garbage cannot leak
        for (int c = 0; c < 4; ++c) {
          int j = lane + (c << 5);
          sAfrag[((i >> 4) * 4 + c) * 512 + frag_slot(i & 15, j & 31)] = (_Float16)0.f;
        }
        continue;
      }
      float si = sSrc[i * 4 + hd];
      float e[4]; float mx = -3e38f;
      for (int c = 0; c < 4; ++c) {
        int j = lane + (c << 5);
        float ev = -3e38f;
        if (j < NROI) {
          float av = sAdj[i * 120 + j];
          float tv = si + sDst[j * 4 + hd];
          tv = tv > 0.f ? tv : 0.2f * tv;                     // leaky_relu 0.2
          ev = (fabsf(av) < 1e-6f) ? -1e9f : tv + 0.5f * av;  // mask + 0.5*adj
        }
        e[c] = ev; mx = fmaxf(mx, ev);
      }
      for (int o = 16; o; o >>= 1) mx = fmaxf(mx, __shfl_xor(mx, o, 32));
      float sum = 0.f;
      for (int c = 0; c < 4; ++c) {
        float zz = (e[c] <= -2e38f) ? 0.f : __expf(e[c] - mx);
        e[c] = zz; sum += zz;
      }
      for (int o = 16; o; o >>= 1) sum += __shfl_xor(sum, o, 32);
      float inv = 1.f / sum;
      for (int c = 0; c < 4; ++c) {
        int j = lane + (c << 5);
        sAfrag[((i >> 4) * 4 + c) * 512 + frag_slot(i & 15, j & 31)] = (_Float16)(e[c] * inv);
      }
    }
    __syncthreads();

    // aggregation: out[:, 32*hd : 32*hd+32] = alpha @ Wh  (2 N-tiles)
    v8f z = {};
    v8f acc2[2] = { z, z };
#pragma unroll
    for (int kt = 0; kt < 4; ++kt) {
      v16h a = *(const v16h*)(sAfrag + (wave * 4 + kt) * 512 + lane * 16);
#pragma unroll
      for (int q = 0; q < 2; ++q) {
        v16h bb = *(const v16h*)(sBfrag + (kt * 8 + hd * 2 + q) * 512 + lane * 16);
        acc2[q] = __builtin_amdgcn_wmma_f32_16x16x32_f16(
            false, a, false, bb, (short)0, acc2[q], false, false);
      }
    }
#pragma unroll
    for (int q = 0; q < 2; ++q)
#pragma unroll
      for (int r = 0; r < 8; ++r) {
        int row = wave * 16 + r + ((lane >> 4) << 3);
        int col = (hd * 2 + q) * 16 + (lane & 15);
        sh[row * CP + col] = acc2[q][r];   // old h is dead; columns disjoint per head
      }
    __syncthreads();
  }

  // ---- epilogue: h_new = elu(bn(out)) + Whf ; zero padded rows
  const float *g = p.gat_g[l], *bb2 = p.gat_b[l], *rm = p.gat_rm[l], *rv = p.gat_rv[l];
  for (int t = tid; t < NCH * NCH; t += NTHR) {
    int i = t >> 7, c = t & 127;
    float v = 0.f;
    if (i < NROI) {
      float o = sh[i * CP + c];
      o = (o - rm[c]) * g[c] * rsqrtf(rv[c] + 1e-5f) + bb2[c];
      v = elu_f(o) + sWh[i * CP + c];
    }
    sh[i * CP + c] = v;
  }
  __syncthreads();
}

__global__ __launch_bounds__(NTHR, 1) void fnpgnn_block(Params p) {
  __shared__ alignas(32) float    sh[NCH * CP];
  __shared__ alignas(32) float    sWh[NCH * CP];
  __shared__ alignas(32) _Float16 sAfrag[8 * 4 * 512];   // [mt][kt][lane][16 halfs]
  __shared__ alignas(32) _Float16 sBfrag[4 * 8 * 512];   // [kt][nt][lane][16 halfs]
  __shared__ float sAdj[NROI * 120];
  __shared__ float sSrc[NCH * 4], sDst[NCH * 4];
  __shared__ float sVn[NCH], sMean[NCH];
  __shared__ float sPooled[9 * NCH];
  __shared__ float sHid[9 * 32];
  __shared__ float sAtt[16];
  __shared__ float sRed[32];

  const int b = blockIdx.x;
  const int tid = threadIdx.x, wave = tid >> 5, lane = tid & 31;

  // ---- adj -> LDS (f32, exact mask semantics)
  const float* adjb = p.adj + (size_t)b * NROI * NROI;
  for (int t = tid; t < NROI * NROI; t += NTHR) {
    int i = t / NROI, j = t - i * NROI;
    sAdj[i * 120 + j] = adjb[t];
  }

  // ---- encoder: stage x as A frags (pad 125->128), enc_W as B frags
  const float* xb = p.x + (size_t)b * NROI * 125;
  for (int t = tid; t < NCH * NCH; t += NTHR) {
    int m = t >> 7, k = t & 127;
    float v = (m < NROI && k < 125) ? xb[m * 125 + k] : 0.f;
    sAfrag[((m >> 4) * 4 + (k >> 5)) * 512 + frag_slot(m & 15, k & 31)] = (_Float16)v;
  }
  for (int t = tid; t < NCH * NCH; t += NTHR) {
    int n = t >> 7, k = t & 127;
    float v = (k < 125) ? p.enc_W[n * 125 + k] : 0.f;
    sBfrag[((k >> 5) * 8 + (n >> 4)) * 512 + frag_slot(n & 15, k & 31)] = (_Float16)v;
  }
  __syncthreads();

  {
    v8f acc[8];
    gemm128(sAfrag, sBfrag, acc);
#pragma unroll
    for (int q = 0; q < 8; ++q)
#pragma unroll
      for (int r = 0; r < 8; ++r) {
        int row = wave * 16 + r + ((lane >> 4) << 3);
        int col = q * 16 + (lane & 15);
        float v = 0.f;
        if (row < NROI) {
          v = acc[q][r] + p.enc_b[col];
          v = elu_f(v);                                                         // elu first
          v = (v - p.bn_in_rm[col]) * p.bn_in_g[col] * rsqrtf(p.bn_in_rv[col] + 1e-5f)
              + p.bn_in_b[col];                                                  // then bn_in
        }
        sh[row * CP + col] = v;
      }
  }
  __syncthreads();

  restageA(sh, sAfrag);
  colmean(sh, sMean);
  __syncthreads();
  if (tid < NCH) sVn[tid] = p.vne[tid] + sMean[tid];
  __syncthreads();

  // ---- gat1 / gat2 (+ virtual-node update), gat3
  for (int l = 0; l < 2; ++l) {
    gat_layer(p, l, sh, sWh, sAfrag, sBfrag, sAdj, sSrc, sDst);
    colmean(sh, sMean);
    __syncthreads();
    float upd = 0.f;
    if (tid < NCH) {
      const float* wr = p.vnu_W + tid * 256;
      float a = p.vnu_b[tid];
      for (int k = 0; k < 128; ++k) a = fmaf(wr[k], sVn[k], a);
      for (int k = 0; k < 128; ++k) a = fmaf(wr[128 + k], sMean[k], a);
      upd = a;
    }
    __syncthreads();
    if (tid < NCH) sVn[tid] = elu_f(upd);
    __syncthreads();
    for (int t = tid; t < NROI * NCH; t += NTHR) {
      int i = t >> 7, c = t & 127;
      sh[i * CP + c] += 0.1f * sVn[c];
    }
    __syncthreads();
    restageA(sh, sAfrag);
    __syncthreads();
  }
  gat_layer(p, 2, sh, sWh, sAfrag, sBfrag, sAdj, sSrc, sDst);

  // ---- network pooling (sum over ROI members)
  for (int t = tid; t < 9 * NCH; t += NTHR) {
    int k = t >> 7, c = t & 127;
    float s = 0.f;
    for (int u = NET_OFF[k]; u < NET_OFF[k + 1]; ++u) s += sh[NET_IDX[u] * CP + c];
    sPooled[t] = s;
  }
  __syncthreads();

  // ---- network attention: tanh(pooled@W1^T+b1)@W2^T+b2, softmax over 9
  for (int t = tid; t < 9 * 32; t += NTHR) {
    int k = t >> 5, j = t & 31;
    const float* w = p.attn_W1 + j * NCH;
    float a = p.attn_b1[j];
    for (int c = 0; c < NCH; ++c) a = fmaf(w[c], sPooled[k * NCH + c], a);
    sHid[t] = tanhf(a);
  }
  __syncthreads();
  if (tid < 9) {
    float s = p.attn_b2[0];
    for (int j = 0; j < 32; ++j) s = fmaf(p.attn_W2[j], sHid[tid * 32 + j], s);
    sAtt[tid] = s;
  }
  __syncthreads();
  if (tid == 0) {
    float mx = -3e38f;
    for (int k = 0; k < 9; ++k) mx = fmaxf(mx, sAtt[k]);
    float s = 0.f;
    for (int k = 0; k < 9; ++k) { float e = __expf(sAtt[k] - mx); sAtt[k] = e; s += e; }
    float inv = 1.f / s;
    for (int k = 0; k < 9; ++k) sAtt[k] *= inv;
  }
  __syncthreads();

  // ---- flat_in = [pooled*a (1152), vn (128)]; LayerNorm; write flat
  float s1 = 0.f, s2 = 0.f;
  for (int idx = tid; idx < 1280; idx += NTHR) {
    float v = (idx < 1152) ? sPooled[idx] * sAtt[idx >> 7] : sVn[idx - 1152];
    s1 += v; s2 += v * v;
  }
  for (int o = 16; o; o >>= 1) { s1 += __shfl_xor(s1, o, 32); s2 += __shfl_xor(s2, o, 32); }
  if (lane == 0) { sRed[wave] = s1; sRed[8 + wave] = s2; }
  __syncthreads();
  if (tid == 0) {
    float a = 0.f, q = 0.f;
    for (int w = 0; w < 8; ++w) { a += sRed[w]; q += sRed[8 + w]; }
    float mu = a * (1.f / 1280.f);
    float var = q * (1.f / 1280.f) - mu * mu;
    sRed[16] = mu; sRed[17] = rsqrtf(var + 1e-5f);
  }
  __syncthreads();
  float mu = sRed[16], rstd = sRed[17];
  float* fo = p.flat_out + (size_t)b * 1280;
  for (int idx = tid; idx < 1280; idx += NTHR) {
    float v = (idx < 1152) ? sPooled[idx] * sAtt[idx >> 7] : sVn[idx - 1152];
    fo[idx] = (v - mu) * rstd * p.ln_g[idx] + p.ln_b[idx];
  }
}

// =====================================================================
// Classifier: logits = elu(flat @ W1^T + b1) @ W2^T + b2
// Blocked WMMA GEMM: M=1024 (64 rows/block), N=256, K=1280.
// =====================================================================
__global__ __launch_bounds__(NTHR, 1) void fnpgnn_cls(const float* __restrict__ flat,
                                                      const float* __restrict__ W1,
                                                      const float* __restrict__ b1,
                                                      const float* __restrict__ W2,
                                                      const float* __restrict__ b2,
                                                      float* __restrict__ logits) {
  __shared__ alignas(32) _Float16 sA[4 * 512];    // 4 M-tiles x one K-step
  __shared__ alignas(32) _Float16 sB[16 * 512];   // 16 N-tiles x one K-step
  __shared__ float sHid[64 * 257];                // padded stride vs bank conflicts

  const int tid = threadIdx.x, wave = tid >> 5, lane = tid & 31;
  const int row0 = blockIdx.x * 64;
  const int mt = wave >> 1, ntbase = (wave & 1) * 8;

  v8f z = {};
  v8f acc[8];
#pragma unroll
  for (int q = 0; q < 8; ++q) acc[q] = z;

  for (int k0 = 0; k0 < 1280; k0 += 32) {
    for (int t = tid; t < 64 * 32; t += NTHR) {
      int m = t >> 5, kk = t & 31;
      sA[(m >> 4) * 512 + frag_slot(m & 15, kk)] = (_Float16)flat[(size_t)(row0 + m) * 1280 + k0 + kk];
    }
    for (int t = tid; t < 256 * 32; t += NTHR) {
      int n = t >> 5, kk = t & 31;
      sB[(n >> 4) * 512 + frag_slot(n & 15, kk)] = (_Float16)W1[(size_t)n * 1280 + k0 + kk];
    }
    __syncthreads();
    v16h a = *(const v16h*)(sA + mt * 512 + lane * 16);
#pragma unroll
    for (int q = 0; q < 8; ++q) {
      v16h bb = *(const v16h*)(sB + (ntbase + q) * 512 + lane * 16);
      acc[q] = __builtin_amdgcn_wmma_f32_16x16x32_f16(
          false, a, false, bb, (short)0, acc[q], false, false);
    }
    __syncthreads();
  }

#pragma unroll
  for (int q = 0; q < 8; ++q)
#pragma unroll
    for (int r = 0; r < 8; ++r) {
      int m = mt * 16 + r + ((lane >> 4) << 3);
      int n = (ntbase + q) * 16 + (lane & 15);
      sHid[m * 257 + n] = elu_f(acc[q][r] + b1[n]);
    }
  __syncthreads();

  for (int t = tid; t < 128; t += NTHR) {
    int m = t >> 1, o = t & 1;
    const float* w = W2 + o * 256;
    float s = b2[o];
    for (int j = 0; j < 256; ++j) s = fmaf(w[j], sHid[m * 257 + j], s);
    logits[(size_t)(row0 + m) * 2 + o] = s;
  }
}

extern "C" void kernel_launch(void* const* d_in, const int* in_sizes, int n_in,
                              void* d_out, int out_size, void* d_ws, size_t ws_size,
                              hipStream_t stream) {
  (void)in_sizes; (void)d_ws; (void)ws_size; (void)out_size;
  if (n_in < 42) return;
  auto in = [&](int i) { return (const float*)d_in[i]; };

  Params p;
  p.x = in(0); p.adj = in(1);
  p.enc_W = in(2); p.enc_b = in(3);
  p.bn_in_g = in(4); p.bn_in_b = in(5); p.bn_in_rm = in(6); p.bn_in_rv = in(7);
  p.vne = in(8);
  for (int l = 0; l < 3; ++l) {
    int base = 9 + 7 * l;
    p.gat_W[l]    = in(base + 0);
    p.gat_asrc[l] = in(base + 1);
    p.gat_adst[l] = in(base + 2);
    p.gat_g[l]    = in(base + 3);
    p.gat_b[l]    = in(base + 4);
    p.gat_rm[l]   = in(base + 5);
    p.gat_rv[l]   = in(base + 6);
  }
  p.vnu_W = in(30); p.vnu_b = in(31);
  p.attn_W1 = in(32); p.attn_b1 = in(33);
  p.attn_W2 = in(34); p.attn_b2 = in(35);
  p.ln_g = in(36); p.ln_b = in(37);

  float* out = (float*)d_out;
  p.flat_out = out + 1024 * 2;

  fnpgnn_block<<<1024, NTHR, 0, stream>>>(p);
  fnpgnn_cls<<<16, NTHR, 0, stream>>>(p.flat_out, in(38), in(39), in(40), in(41), out);
}
